// Net3DDistancePredictor_44530220924978
// MI455X (gfx1250) — matrix operand order repaired and checked by reading, not compiled, not run on hardware
//
#include <hip/hip_runtime.h>
#include <hip/hip_bf16.h>

// Net3D distance predictor for MI455X (gfx1250, wave32, WMMA).
// Edge MLP dominates (~420 GFLOP) -> bf16 WMMA, M=32 edges/wave to amortize
// B-fragment traffic (W1+W2 = 128KB read once per 32 edges => ~32 FLOP/B vs L2).
// Weights pre-swizzled lane-major so every B fragment is 2x global_load_b128.

#define NN 50000
#define EE 800000
#define PP 500000
#define HH 128
#define LL 4

typedef __attribute__((ext_vector_type(16))) __bf16       v16bf;
typedef __attribute__((ext_vector_type(8)))  float        v8f;
typedef __attribute__((ext_vector_type(8)))  unsigned int v8u;
typedef __attribute__((ext_vector_type(4)))  unsigned int v4u;

__device__ __forceinline__ unsigned pk_bf16(float lo, float hi) {
  unsigned d;
  asm("v_cvt_pk_bf16_f32 %0, %1, %2" : "=v"(d) : "v"(lo), "v"(hi));
  return d;
}
__device__ __forceinline__ float siluf(float x) { return x / (1.f + __expf(-x)); }
__device__ __forceinline__ float sigmf(float x) { return 1.f / (1.f + __expf(-x)); }

__device__ __forceinline__ v8f wmma_bf16(v8u a, v8u b, v8f c) {
  return __builtin_amdgcn_wmma_f32_16x16x32_bf16(
      false, __builtin_bit_cast(v16bf, a),
      false, __builtin_bit_cast(v16bf, b),
      (short)0, c, false, false);
}

__device__ __forceinline__ v8u join8(v4u a, v4u b) {
  v8u f;
  f[0] = a[0]; f[1] = a[1]; f[2] = a[2]; f[3] = a[3];
  f[4] = b[0]; f[5] = b[1]; f[6] = b[2]; f[7] = b[3];
  return f;
}

// A-matrix 16x32 bf16: lane (half=lane>>4) holds K = kbase+8*half+0..7 (VGPR 0-3)
// and K = kbase+16+8*half+0..7 (VGPR 4-7) -> two contiguous 8-float runs.
__device__ __forceinline__ v8u a_frag_g(const float* rowp, int kbase, int half) {
  const float4* p = (const float4*)(rowp + kbase + 8 * half);
  const float4* q = (const float4*)(rowp + kbase + 16 + 8 * half);
  float4 a = p[0], b = p[1], c = q[0], d = q[1];
  v8u f;
  f[0] = pk_bf16(a.x, a.y); f[1] = pk_bf16(a.z, a.w);
  f[2] = pk_bf16(b.x, b.y); f[3] = pk_bf16(b.z, b.w);
  f[4] = pk_bf16(c.x, c.y); f[5] = pk_bf16(c.z, c.w);
  f[6] = pk_bf16(d.x, d.y); f[7] = pk_bf16(d.z, d.w);
  return f;
}
__device__ __forceinline__ v8u a_frag_sum_g(const float* r1, const float* r2, int kbase, int half) {
  const float4* p1 = (const float4*)(r1 + kbase + 8 * half);
  const float4* q1 = (const float4*)(r1 + kbase + 16 + 8 * half);
  const float4* p2 = (const float4*)(r2 + kbase + 8 * half);
  const float4* q2 = (const float4*)(r2 + kbase + 16 + 8 * half);
  float4 a = p1[0], b = p1[1], c = q1[0], d = q1[1];
  float4 aa = p2[0], bb = p2[1], cc = q2[0], dd = q2[1];
  v8u f;
  f[0] = pk_bf16(a.x + aa.x, a.y + aa.y); f[1] = pk_bf16(a.z + aa.z, a.w + aa.w);
  f[2] = pk_bf16(b.x + bb.x, b.y + bb.y); f[3] = pk_bf16(b.z + bb.z, b.w + bb.w);
  f[4] = pk_bf16(c.x + cc.x, c.y + cc.y); f[5] = pk_bf16(c.z + cc.z, c.w + cc.w);
  f[6] = pk_bf16(d.x + dd.x, d.y + dd.y); f[7] = pk_bf16(d.z + dd.z, d.w + dd.w);
  return f;
}
__device__ __forceinline__ v8u a_frag_lds(const unsigned short* sm, int row, int kbase, int half) {
  const v4u* p = (const v4u*)(sm + row * HH + kbase + 8 * half);
  const v4u* q = (const v4u*)(sm + row * HH + kbase + 16 + 8 * half);
  return join8(p[0], q[0]);
}
// B fragment: lane-major pre-swizzled, 8 contiguous dwords per lane -> 2x b128.
__device__ __forceinline__ v8u b_frag(const unsigned* tilebase, int lane) {
  const v4u* p = (const v4u*)(tilebase + lane * 8);
  return join8(p[0], p[1]);
}

// ---- weight pre-swizzle: W[K x N] f32 row-major -> bf16 B fragments, lane-major ----
__global__ void k_swz(const float* __restrict__ W, unsigned* __restrict__ out, int K, int N) {
  int tid = blockIdx.x * blockDim.x + threadIdx.x;
  int total = (K * N) >> 1;
  if (tid >= total) return;
  int i = tid & 7, lane = (tid >> 3) & 31, tile = tid >> 8;
  int NT = N >> 4;
  int kt = tile / NT, nt = tile - kt * NT;
  int half = lane >> 4, n = lane & 15;
  int k0 = kt * 32 + 8 * half + 2 * (i & 3) + 16 * (i >> 2);    // B layout: lane half h holds K 8h..8h+7, 16+8h..16+8h+7
  int col = nt * 16 + n;
  out[tid] = pk_bf16(W[(size_t)k0 * N + col], W[(size_t)(k0 + 1) * N + col]);
}

__global__ void k_init_h(const float* __restrict__ emb, float* __restrict__ h) {
  int tid = blockIdx.x * 256 + threadIdx.x;          // NN*32 threads
  int c4 = (tid & 31) * 4;
  size_t nd = (size_t)(tid >> 5);
  *(float4*)(h + nd * HH + c4) = *(const float4*)(emb + c4);
}

__global__ void k_edge_init(const float* __restrict__ ed, const float* __restrict__ We,
                            const float* __restrict__ be, float* __restrict__ e) {
  int tid = blockIdx.x * 256 + threadIdx.x;          // EE*32 threads
  int c4 = (tid & 31) * 4;
  size_t eg = (size_t)(tid >> 5);
  float d = ed[eg];
  float4 w = *(const float4*)(We + c4);
  float4 b = *(const float4*)(be + c4);
  float4 o;
  o.x = siluf(siluf(d * w.x + b.x));
  o.y = siluf(siluf(d * w.y + b.y));
  o.z = siluf(siluf(d * w.z + b.z));
  o.w = siluf(siluf(d * w.w + b.w));
  *(float4*)(e + eg * HH + c4) = o;
}

__global__ void k_zero4(float4* __restrict__ p) {
  p[(size_t)blockIdx.x * 256 + threadIdx.x] = make_float4(0.f, 0.f, 0.f, 0.f);
}

// ---- message kernel: 1 wave = 32 edges (2 subtiles share every B fragment) ----
__global__ __launch_bounds__(128) void k_msg(
    const float* __restrict__ h, float* __restrict__ e,
    const int* __restrict__ src, const int* __restrict__ dst,
    const unsigned* __restrict__ w1s, const float* __restrict__ b1,
    const unsigned* __restrict__ w2s, const float* __restrict__ b2,
    const float* __restrict__ sw, const float* __restrict__ sb,
    float* msum) {
  __shared__ unsigned short sInt[4][32 * HH];        // 4 waves x 8KB
  const int lane = threadIdx.x & 31;
  const int wave = threadIdx.x >> 5;
  const int t32 = blockIdx.x * 4 + wave;
  const int half = lane >> 4, n = lane & 15, row = lane & 15;
  const int eg0 = t32 * 32 + row, eg1 = eg0 + 16;
  const float* hs0 = h + (size_t)src[eg0] * HH;
  const float* hd0 = h + (size_t)dst[eg0] * HH;
  const float* er0 = e + (size_t)eg0 * HH;
  const float* hs1 = h + (size_t)src[eg1] * HH;
  const float* hd1 = h + (size_t)dst[eg1] * HH;
  const float* er1 = e + (size_t)eg1 * HH;

  v8f acc[2][8];
#pragma unroll
  for (int nt = 0; nt < 8; ++nt) {
    float bb = b1[nt * 16 + n];
#pragma unroll
    for (int j = 0; j < 8; ++j) { acc[0][nt][j] = bb; acc[1][nt][j] = bb; }
  }

  // GEMM1: [32 x 384] @ [384 x 128]; mi = [h_src | h_dst | e]
#pragma unroll 1
  for (int kt = 0; kt < 12; ++kt) {
    const float* r0 = (kt < 4) ? hs0 : ((kt < 8) ? hd0 : er0);
    const float* r1 = (kt < 4) ? hs1 : ((kt < 8) ? hd1 : er1);
    int kb = (kt & 3) * 32;
    v8u A0 = a_frag_g(r0, kb, half);
    v8u A1 = a_frag_g(r1, kb, half);
    const unsigned* wb = w1s + (size_t)kt * 8 * 256;
#pragma unroll
    for (int nt = 0; nt < 8; ++nt) {
      v8u B = b_frag(wb + nt * 256, lane);
      acc[0][nt] = wmma_bf16(A0, B, acc[0][nt]);
      acc[1][nt] = wmma_bf16(A1, B, acc[1][nt]);
    }
  }

  unsigned short* sm = sInt[wave];
#pragma unroll
  for (int s = 0; s < 2; ++s)
#pragma unroll
    for (int nt = 0; nt < 8; ++nt)
#pragma unroll
      for (int j = 0; j < 8; ++j)
        sm[(s * 16 + j + 8 * half) * HH + nt * 16 + n] =
            (unsigned short)pk_bf16(siluf(acc[s][nt][j]), 0.f);
  __syncthreads();

  // GEMM2: [32 x 128] @ [128 x 128]
#pragma unroll
  for (int nt = 0; nt < 8; ++nt) {
    float bb = b2[nt * 16 + n];
#pragma unroll
    for (int j = 0; j < 8; ++j) { acc[0][nt][j] = bb; acc[1][nt][j] = bb; }
  }
#pragma unroll 1
  for (int kt = 0; kt < 4; ++kt) {
    v8u A0 = a_frag_lds(sm, row, kt * 32, half);
    v8u A1 = a_frag_lds(sm, 16 + row, kt * 32, half);
    const unsigned* wb = w2s + (size_t)kt * 8 * 256;
#pragma unroll
    for (int nt = 0; nt < 8; ++nt) {
      v8u B = b_frag(wb + nt * 256, lane);
      acc[0][nt] = wmma_bf16(A0, B, acc[0][nt]);
      acc[1][nt] = wmma_bf16(A1, B, acc[1][nt]);
    }
  }
#pragma unroll
  for (int s = 0; s < 2; ++s)
#pragma unroll
    for (int nt = 0; nt < 8; ++nt)
#pragma unroll
      for (int j = 0; j < 8; ++j) acc[s][nt][j] = siluf(acc[s][nt][j]);

  // gate = sigmoid(m @ soft_w + soft_b); reduce over 16-lane halves
  float sbv = sb[0];
  float gate[2][8];
#pragma unroll
  for (int s = 0; s < 2; ++s) {
    float gp[8];
#pragma unroll
    for (int j = 0; j < 8; ++j) gp[j] = 0.f;
#pragma unroll
    for (int nt = 0; nt < 8; ++nt) {
      float w = sw[nt * 16 + n];
#pragma unroll
      for (int j = 0; j < 8; ++j) gp[j] += acc[s][nt][j] * w;
    }
#pragma unroll
    for (int msk = 8; msk >= 1; msk >>= 1)
#pragma unroll
      for (int j = 0; j < 8; ++j) gp[j] += __shfl_xor(gp[j], msk, 32);
#pragma unroll
    for (int j = 0; j < 8; ++j) gate[s][j] = sigmf(gp[j] + sbv);
  }

  // e += m ; scatter m*gate into msum[dst] (fully unrolled: acc stays in VGPRs)
#pragma unroll
  for (int s = 0; s < 2; ++s) {
    int dr[8];
#pragma unroll
    for (int j = 0; j < 8; ++j) dr[j] = dst[t32 * 32 + s * 16 + j + 8 * half];
#pragma unroll
    for (int nt = 0; nt < 8; ++nt) {
      int c = nt * 16 + n;
#pragma unroll
      for (int j = 0; j < 8; ++j) {
        float mv = acc[s][nt][j];
        e[(size_t)(t32 * 32 + s * 16 + j + 8 * half) * HH + c] += mv;
        __hip_atomic_fetch_add(&msum[(size_t)dr[j] * HH + c], mv * gate[s][j],
                               __ATOMIC_RELAXED, __HIP_MEMORY_SCOPE_AGENT);
      }
    }
  }
}

// ---- node update kernel: 1 wave = 16 nodes ----
__global__ __launch_bounds__(32) void k_upd(
    float* __restrict__ h, const float* __restrict__ msum,
    const unsigned* __restrict__ w1s, const float* __restrict__ b1,
    const unsigned* __restrict__ w2s, const float* __restrict__ b2) {
  __shared__ unsigned short sm[16 * HH];
  const int lane = threadIdx.x & 31;
  const int tile = blockIdx.x;
  const int half = lane >> 4, n = lane & 15, row = lane & 15;
  const float* hr = h + (size_t)(tile * 16 + row) * HH;
  const float* mr = msum + (size_t)(tile * 16 + row) * HH;

  v8f acc[8];
#pragma unroll
  for (int nt = 0; nt < 8; ++nt) {
    float bb = b1[nt * 16 + n];
#pragma unroll
    for (int j = 0; j < 8; ++j) acc[nt][j] = bb;
  }
#pragma unroll 1
  for (int kt = 0; kt < 4; ++kt) {
    v8u A = a_frag_sum_g(hr, mr, kt * 32, half);
    const unsigned* wb = w1s + (size_t)kt * 8 * 256;
#pragma unroll
    for (int nt = 0; nt < 8; ++nt)
      acc[nt] = wmma_bf16(A, b_frag(wb + nt * 256, lane), acc[nt]);
  }
#pragma unroll
  for (int nt = 0; nt < 8; ++nt)
#pragma unroll
    for (int j = 0; j < 8; ++j)
      sm[(j + 8 * half) * HH + nt * 16 + n] =
          (unsigned short)pk_bf16(siluf(acc[nt][j]), 0.f);
  __syncthreads();
#pragma unroll
  for (int nt = 0; nt < 8; ++nt) {
    float bb = b2[nt * 16 + n];
#pragma unroll
    for (int j = 0; j < 8; ++j) acc[nt][j] = bb;
  }
#pragma unroll 1
  for (int kt = 0; kt < 4; ++kt) {
    v8u A = a_frag_lds(sm, row, kt * 32, half);
    const unsigned* wb = w2s + (size_t)kt * 8 * 256;
#pragma unroll
    for (int nt = 0; nt < 8; ++nt)
      acc[nt] = wmma_bf16(A, b_frag(wb + nt * 256, lane), acc[nt]);
  }
#pragma unroll
  for (int nt = 0; nt < 8; ++nt)
#pragma unroll
    for (int j = 0; j < 8; ++j)
      h[(size_t)(tile * 16 + j + 8 * half) * HH + nt * 16 + n] += acc[nt][j];
}

// ---- distance head: softplus((h[i]+h[j]) . (Wd[:H]+Wd[H:]) + 2*bd) ----
__global__ __launch_bounds__(256) void k_dist(
    const float* __restrict__ h, const int* __restrict__ pidx,
    const float* __restrict__ Wd, const float* __restrict__ bd,
    float* __restrict__ out) {
  const int lane = threadIdx.x & 31;
  const int p = blockIdx.x * 8 + (threadIdx.x >> 5);
  int a = pidx[p], b = pidx[PP + p];
  const float4* ha = (const float4*)(h + (size_t)a * HH);
  const float4* hb = (const float4*)(h + (size_t)b * HH);
  const float4* w4 = (const float4*)Wd;
  float4 s = ha[lane], t = hb[lane];
  float4 wa = w4[lane], wb = w4[32 + lane];
  float dot = (s.x + t.x) * (wa.x + wb.x) + (s.y + t.y) * (wa.y + wb.y) +
              (s.z + t.z) * (wa.z + wb.z) + (s.w + t.w) * (wa.w + wb.w);
#pragma unroll
  for (int msk = 16; msk >= 1; msk >>= 1) dot += __shfl_xor(dot, msk, 32);
  if (lane == 0) {
    float z = dot + 2.f * bd[0];
    out[p] = (z > 20.f) ? z : log1pf(__expf(z));
  }
}

extern "C" void kernel_launch(void* const* d_in, const int* in_sizes, int n_in,
                              void* d_out, int out_size, void* d_ws, size_t ws_size,
                              hipStream_t stream) {
  (void)in_sizes; (void)n_in; (void)out_size; (void)ws_size;
  const float* edge_d = (const float*)d_in[0];
  const float* emb    = (const float*)d_in[1];
  const float* We     = (const float*)d_in[2];
  const float* be     = (const float*)d_in[3];
  const float* msgW1  = (const float*)d_in[4];
  const float* msgb1  = (const float*)d_in[5];
  const float* msgW2  = (const float*)d_in[6];
  const float* msgb2  = (const float*)d_in[7];
  const float* softw  = (const float*)d_in[8];
  const float* softb  = (const float*)d_in[9];
  const float* updW1  = (const float*)d_in[10];
  const float* updb1  = (const float*)d_in[11];
  const float* updW2  = (const float*)d_in[12];
  const float* updb2  = (const float*)d_in[13];
  const float* Wd     = (const float*)d_in[14];
  const float* bd     = (const float*)d_in[15];
  const int*   src    = (const int*)d_in[16];
  const int*   dst    = (const int*)d_in[17];
  const int*   pidx   = (const int*)d_in[18];
  float* out = (float*)d_out;

  char* ws = (char*)d_ws;
  size_t off = 0;
  float* e    = (float*)(ws + off); off += (size_t)EE * HH * sizeof(float);   // 409.6 MB
  float* hbuf = (float*)(ws + off); off += (size_t)NN * HH * sizeof(float);   // 25.6 MB
  float* ms   = (float*)(ws + off); off += (size_t)NN * HH * sizeof(float);   // 25.6 MB
  unsigned* w1s = (unsigned*)(ws + off); off += (size_t)LL * 24576 * 4;
  unsigned* w2s = (unsigned*)(ws + off); off += (size_t)LL * 8192 * 4;
  unsigned* u1s = (unsigned*)(ws + off); off += (size_t)LL * 8192 * 4;
  unsigned* u2s = (unsigned*)(ws + off); off += (size_t)LL * 8192 * 4;

  for (int l = 0; l < LL; ++l) {
    k_swz<<<96, 256, 0, stream>>>(msgW1 + (size_t)l * 384 * HH, w1s + (size_t)l * 24576, 384, HH);
    k_swz<<<32, 256, 0, stream>>>(msgW2 + (size_t)l * HH * HH,  w2s + (size_t)l * 8192,  HH, HH);
    k_swz<<<32, 256, 0, stream>>>(updW1 + (size_t)l * HH * HH,  u1s + (size_t)l * 8192,  HH, HH);
    k_swz<<<32, 256, 0, stream>>>(updW2 + (size_t)l * HH * HH,  u2s + (size_t)l * 8192,  HH, HH);
  }
  k_init_h<<<NN * 32 / 256, 256, 0, stream>>>(emb, hbuf);
  k_edge_init<<<EE * 32 / 256, 256, 0, stream>>>(edge_d, We, be, e);

  for (int l = 0; l < LL; ++l) {
    k_zero4<<<NN * HH / 4 / 256, 256, 0, stream>>>((float4*)ms);
    k_msg<<<EE / 128, 128, 0, stream>>>(hbuf, e, src, dst,
        w1s + (size_t)l * 24576, msgb1 + (size_t)l * HH,
        w2s + (size_t)l * 8192,  msgb2 + (size_t)l * HH,
        softw + (size_t)l * HH, softb + l, ms);
    k_upd<<<NN / 16, 32, 0, stream>>>(hbuf, ms,
        u1s + (size_t)l * 8192, updb1 + (size_t)l * HH,
        u2s + (size_t)l * 8192, updb2 + (size_t)l * HH);
  }
  k_dist<<<PP / 8, 256, 0, stream>>>(hbuf, pidx, Wd, bd, out);
}
